// IndexEmbedder_57208964382808
// MI455X (gfx1250) — matrix operand order repaired and checked by reading, not compiled, not run on hardware
//
#include <hip/hip_runtime.h>

// ---------------------------------------------------------------------------
// Problem constants (match reference)
// ---------------------------------------------------------------------------
#define B_Q     2048
#define N_A     65536
#define DIM     768
#define DP2     (DIM / 2)       // 384 packed bf16-pairs (uint32) per row
#define KFRAG   (DIM / 32)      // 24 WMMA K-steps
#define KSEL    8               // top_k * 2
#define QTILE   128             // queries per workgroup (8 bands of 16)
#define NCHUNK  16              // anchors per inner chunk (shared by all waves)
#define NSPLIT  32              // anchor-range splits per query tile
#define THREADS 256             // 8 wave32 per block
#define CHUNKU  (NCHUNK * DP2)  // uints per staged chunk (6144 = 24 KiB)

typedef __attribute__((ext_vector_type(16))) __bf16 v16bf;
typedef __attribute__((ext_vector_type(8)))  float  v8f;
typedef __attribute__((ext_vector_type(4)))  int    v4i_t;

#if __has_builtin(__builtin_amdgcn_global_load_async_to_lds_b128)
#define HAVE_ASYNC_LDS 1
#else
#define HAVE_ASYNC_LDS 0
#endif

// fp32 -> bf16 (round-to-nearest-even), packed pair
__device__ __forceinline__ unsigned int pack_bf16x2(float lo, float hi) {
    unsigned int ulo = __float_as_uint(lo);
    unsigned int uhi = __float_as_uint(hi);
    ulo = (ulo + 0x7FFFu + ((ulo >> 16) & 1u)) >> 16;
    uhi = (uhi + 0x7FFFu + ((uhi >> 16) & 1u)) >> 16;
    return ulo | (uhi << 16);
}

// Stage one 16-anchor chunk (24 KiB) global -> LDS, spread over the block.
// Async path: GLOBAL_LOAD_ASYNC_TO_LDS_B128 (ASYNCcnt); fallback: uint4 copy.
__device__ __forceinline__ void stage_chunk(const unsigned int* __restrict__ src,
                                            unsigned int* dstLds, int t) {
#if HAVE_ASYNC_LDS
    typedef __attribute__((address_space(1))) v4i_t g_v4i;
    typedef __attribute__((address_space(3))) v4i_t l_v4i;
    for (int i = t; i < CHUNKU / 4; i += THREADS) {
        g_v4i* gp = (g_v4i*)(size_t)(src + (size_t)i * 4);
        // low 32 bits of a generic LDS address are the LDS offset
        l_v4i* lp = (l_v4i*)(unsigned int)(size_t)(dstLds + (size_t)i * 4);
        __builtin_amdgcn_global_load_async_to_lds_b128(gp, lp, 0, 0);
    }
#else
    for (int i = t; i < CHUNKU / 4; i += THREADS)
        ((uint4*)dstLds)[i] = ((const uint4*)src)[i];
#endif
}

__device__ __forceinline__ void wait_async_done() {
#if HAVE_ASYNC_LDS
#if __has_builtin(__builtin_amdgcn_s_wait_asynccnt)
    __builtin_amdgcn_s_wait_asynccnt(0);
#else
    asm volatile("s_wait_asynccnt 0x0" ::: "memory");
#endif
#endif
}

// ---------------------------------------------------------------------------
// Kernel 1: L2-normalize rows, emit packed bf16
// ---------------------------------------------------------------------------
__global__ void __launch_bounds__(THREADS)
normalize_bf16_kernel(const float* __restrict__ queries,
                      const float* __restrict__ anchors,
                      unsigned int* __restrict__ qn,
                      unsigned int* __restrict__ an) {
    __shared__ float red[THREADS];
    const int row = blockIdx.x;
    const int t   = threadIdx.x;

    const float* src;
    unsigned int* dst;
    if (row < B_Q) {
        src = queries + (size_t)row * DIM;
        dst = qn + (size_t)row * DP2;
    } else {
        const int r = row - B_Q;
        src = anchors + (size_t)r * DIM;
        dst = an + (size_t)r * DP2;
    }

    float ss = 0.f;
    for (int i = t; i < DIM; i += THREADS) {
        float v = src[i];
        ss += v * v;
    }
    red[t] = ss;
    __syncthreads();
    for (int off = THREADS / 2; off > 0; off >>= 1) {
        if (t < off) red[t] += red[t + off];
        __syncthreads();
    }
    const float scale = 1.0f / fmaxf(sqrtf(red[0]), 1e-8f);

    for (int p = t; p < DP2; p += THREADS)
        dst[p] = pack_bf16x2(src[2 * p] * scale, src[2 * p + 1] * scale);
}

// ---------------------------------------------------------------------------
// Kernel 2: bf16-WMMA GEMM (A resident in VGPRs) + per-query partial top-8
// grid: (B_Q/QTILE) * NSPLIT blocks of 256 threads (8 waves)
//   wave = 16-query band; all waves share one 16-anchor LDS chunk (dbl-buffered)
// ---------------------------------------------------------------------------
__global__ void __launch_bounds__(THREADS, 1)
topk_gemm_kernel(const unsigned int* __restrict__ qn,
                 const unsigned int* __restrict__ an,
                 float* __restrict__ pscore,   // [B_Q][NSPLIT][KSEL]
                 int*   __restrict__ pidx) {   // [B_Q][NSPLIT][KSEL]
    extern __shared__ __align__(16) char smem_raw[];
    unsigned int* s_a      = (unsigned int*)smem_raw;       // 2 * CHUNKU u32
    float*        s_scores = (float*)(s_a + 2 * CHUNKU);    // QTILE*NCHUNK f32

    const int t    = threadIdx.x;
    const int lane = t & 31;
    const int band = t >> 5;              // wave = query band (0..7)

    const int nTiles = B_Q / QTILE;       // 16
    const int qTile  = blockIdx.x % nTiles;
    const int split  = blockIdx.x / nTiles;
    const int qBase  = qTile * QTILE;
    const int nLo    = split * (N_A / NSPLIT);
    const int nChunks = (N_A / NSPLIT) / NCHUNK;

    // CDNA5 16-bit A 16x32 layout: lane half selects K {0-7,16-23} vs {8-15,24-31}
    const int aM    = lane & 15;
    const int aHalf = lane >> 4;
    // B 32x16 layout: lane half selects K {0-7,16-23} vs {8-15,24-31}
    const int bN     = lane & 15;
    const int bKpair = (lane >> 4) * 4;

    // ---- Load this wave's full A set (16 queries x K=768) into VGPRs ------
    v16bf afrag[KFRAG];
    {
        const unsigned int* qrow = qn + (size_t)(qBase + band * 16 + aM) * DP2;
#pragma unroll
        for (int f = 0; f < KFRAG; ++f) {
            union { unsigned int u[8]; v16bf v; } fa;
            uint4 a0 = *(const uint4*)(qrow + f * 16 + aHalf * 4);
            uint4 a1 = *(const uint4*)(qrow + f * 16 + 8 + aHalf * 4);
            fa.u[0] = a0.x; fa.u[1] = a0.y; fa.u[2] = a0.z; fa.u[3] = a0.w;
            fa.u[4] = a1.x; fa.u[5] = a1.y; fa.u[6] = a1.z; fa.u[7] = a1.w;
            afrag[f] = fa.v;
        }
    }

    float ts[KSEL];
    int   ti[KSEL];
#pragma unroll
    for (int j = 0; j < KSEL; ++j) { ts[j] = -3.0e38f; ti[j] = 0; }

    // ---- Prologue: stage chunk 0 into buffer 0 -----------------------------
    int cur = 0;
    stage_chunk(an + (size_t)nLo * DP2, s_a, t);
    wait_async_done();
    __syncthreads();

    for (int c = 0; c < nChunks; ++c) {
        const int nBase = nLo + c * NCHUNK;

        // Kick off async staging of the next chunk into the other buffer.
        if (c + 1 < nChunks) {
            stage_chunk(an + (size_t)(nBase + NCHUNK) * DP2,
                        s_a + (cur ^ 1) * CHUNKU, t);
            if (c + 2 < nChunks)   // nudge chunk c+2 toward GL2
                __builtin_prefetch(an + (size_t)(nBase + 2 * NCHUNK) * DP2 + t * 16, 0, 1);
        }

        // ---- 24 x WMMA over K=768, A from registers, B from LDS -----------
        // Software-pipelined B fragments: loads for step f+1 are issued
        // before the WMMA of step f consumes the current buffer, so each
        // WMMA overlaps the next pair of ds_load_b128.
        v8f cacc = {};
        const unsigned int* bRow = s_a + cur * CHUNKU + (size_t)bN * DP2;

        union FB { unsigned int u[8]; v16bf v; };
        FB fb0, fb1;
        {
            uint4 b0 = *(const uint4*)(bRow + bKpair);
            uint4 b1 = *(const uint4*)(bRow + bKpair + 8);
            fb0.u[0] = b0.x; fb0.u[1] = b0.y; fb0.u[2] = b0.z; fb0.u[3] = b0.w;
            fb0.u[4] = b1.x; fb0.u[5] = b1.y; fb0.u[6] = b1.z; fb0.u[7] = b1.w;
        }
#pragma unroll
        for (int f = 0; f < KFRAG; ++f) {
            FB& fcur = (f & 1) ? fb1 : fb0;
            FB& fnxt = (f & 1) ? fb0 : fb1;
            if (f + 1 < KFRAG) {
                uint4 b0 = *(const uint4*)(bRow + (f + 1) * 16 + bKpair);
                uint4 b1 = *(const uint4*)(bRow + (f + 1) * 16 + bKpair + 8);
                fnxt.u[0] = b0.x; fnxt.u[1] = b0.y; fnxt.u[2] = b0.z; fnxt.u[3] = b0.w;
                fnxt.u[4] = b1.x; fnxt.u[5] = b1.y; fnxt.u[6] = b1.z; fnxt.u[7] = b1.w;
            }
            cacc = __builtin_amdgcn_wmma_f32_16x16x32_bf16(
                       false, afrag[f], false, fcur.v, (short)0, cacc, false, false);
        }

        // Scatter 16x16 f32 tile: VGPR r -> M = r + 8*(lane>>4), N = lane&15.
#pragma unroll
        for (int r = 0; r < 8; ++r) {
            const int m = r + 8 * (lane >> 4);
            s_scores[(band * 16 + m) * NCHUNK + (lane & 15)] = cacc[r];
        }
        __syncthreads();

        // Per-query running top-8 (thread t owns query t of the tile).
        if (t < QTILE) {
            const float* rowS = s_scores + t * NCHUNK;
            for (int j = 0; j < NCHUNK; ++j) {
                const float s = rowS[j];
                if (s > ts[KSEL - 1]) {
                    ts[KSEL - 1] = s;
                    ti[KSEL - 1] = nBase + j;
#pragma unroll
                    for (int p = KSEL - 1; p > 0; --p) {
                        if (ts[p] > ts[p - 1]) {
                            float tf = ts[p]; ts[p] = ts[p - 1]; ts[p - 1] = tf;
                            int   tn = ti[p]; ti[p] = ti[p - 1]; ti[p - 1] = tn;
                        }
                    }
                }
            }
        }

        // Next chunk's staging must be complete and visible before reuse.
        wait_async_done();
        __syncthreads();
        cur ^= 1;
    }

    if (t < QTILE) {
        const size_t base = ((size_t)(qBase + t) * NSPLIT + split) * KSEL;
#pragma unroll
        for (int j = 0; j < KSEL; ++j) {
            pscore[base + j] = ts[j];
            pidx[base + j]   = ti[j];
        }
    }
}

// ---------------------------------------------------------------------------
// Kernel 3: merge NSPLIT partial top-8 lists, write outputs, gather anchors
// ---------------------------------------------------------------------------
__global__ void __launch_bounds__(THREADS)
topk_reduce_kernel(const float* __restrict__ anchors_f32,
                   const float* __restrict__ pscore,
                   const int*   __restrict__ pidx,
                   float* __restrict__ out_scores,
                   float* __restrict__ out_idx,
                   float* __restrict__ out_gather) {
    __shared__ int s_idx[THREADS * KSEL];
    const int t = threadIdx.x;
    const int q = blockIdx.x * THREADS + t;

    float ts[KSEL];
    int   ti[KSEL];
#pragma unroll
    for (int j = 0; j < KSEL; ++j) { ts[j] = -3.0e38f; ti[j] = 0; }

    const float* ps = pscore + (size_t)q * NSPLIT * KSEL;
    const int*   pi = pidx   + (size_t)q * NSPLIT * KSEL;
    for (int j = 0; j < NSPLIT * KSEL; ++j) {
        const float s = ps[j];
        if (s > ts[KSEL - 1]) {
            ts[KSEL - 1] = s;
            ti[KSEL - 1] = pi[j];
#pragma unroll
            for (int p = KSEL - 1; p > 0; --p) {
                if (ts[p] > ts[p - 1]) {
                    float tf = ts[p]; ts[p] = ts[p - 1]; ts[p - 1] = tf;
                    int   tn = ti[p]; ti[p] = ti[p - 1]; ti[p - 1] = tn;
                }
            }
        }
    }

#pragma unroll
    for (int j = 0; j < KSEL; ++j) {
        out_scores[(size_t)q * KSEL + j] = ts[j];
        out_idx[(size_t)q * KSEL + j]    = (float)ti[j];
        s_idx[t * KSEL + j]              = ti[j];
    }
    __syncthreads();

    // Cooperative gather of the selected fp32 anchor rows (float4 copies).
    for (int rr = t; rr < THREADS * KSEL; rr += THREADS) {
        const int qq  = rr >> 3;
        const int jj  = rr & (KSEL - 1);
        const int idx = s_idx[rr];
        const float4* src = (const float4*)(anchors_f32 + (size_t)idx * DIM);
        float4* dst = (float4*)(out_gather +
                       ((size_t)(blockIdx.x * THREADS + qq) * KSEL + jj) * DIM);
        for (int e = 0; e < DIM / 4; ++e) dst[e] = src[e];
    }
}

// ---------------------------------------------------------------------------
// Host-side launch
// ---------------------------------------------------------------------------
extern "C" void kernel_launch(void* const* d_in, const int* in_sizes, int n_in,
                              void* d_out, int out_size, void* d_ws, size_t ws_size,
                              hipStream_t stream) {
    (void)in_sizes; (void)n_in; (void)out_size; (void)ws_size;

    const float* queries = (const float*)d_in[0]; // [B_Q, DIM] f32
    const float* anchors = (const float*)d_in[1]; // [N_A, DIM] f32
    // d_in[2] = top_k (device scalar); KSEL = 2*top_k hardcoded per reference.

    // Workspace layout
    unsigned int* qn = (unsigned int*)d_ws;                    // B_Q*DP2 u32
    unsigned int* an = qn + (size_t)B_Q * DP2;                 // N_A*DP2 u32
    float* pscore    = (float*)(an + (size_t)N_A * DP2);       // B_Q*NSPLIT*KSEL
    int*   pidx      = (int*)(pscore + (size_t)B_Q * NSPLIT * KSEL);

    // Output layout (tuple concatenated flat, all as f32)
    float* out_scores = (float*)d_out;                         // [B_Q, KSEL]
    float* out_idx    = out_scores + (size_t)B_Q * KSEL;       // [B_Q, KSEL]
    float* out_gather = out_idx + (size_t)B_Q * KSEL;          // [B_Q, KSEL, DIM]

    normalize_bf16_kernel<<<B_Q + N_A, THREADS, 0, stream>>>(queries, anchors, qn, an);

    const size_t smem = (size_t)(2 * CHUNKU) * sizeof(unsigned int)
                      + (size_t)QTILE * NCHUNK * sizeof(float);
    topk_gemm_kernel<<<(B_Q / QTILE) * NSPLIT, THREADS, smem, stream>>>(
        qn, an, pscore, pidx);

    topk_reduce_kernel<<<B_Q / THREADS, THREADS, 0, stream>>>(
        anchors, pscore, pidx, out_scores, out_idx, out_gather);
}
